// CausalSelfAttention_53283364274521
// MI455X (gfx1250) — compile-verified, matrix operand order checked by reading, not run on hardware
//
#include <hip/hip_runtime.h>
#include <stdint.h>

// ---------------------------------------------------------------------------
// CDNA5 (gfx1250) GQA causal self-attention.
// bf16 WMMA for all matmuls; double-buffered TDM (tensor_load_to_lds) stages
// GEMM panels so DMA overlaps WMMA.
// B=2, T=2048, C=4096, 32 Q heads / 8 KV groups, D=128.
// ---------------------------------------------------------------------------

typedef __attribute__((ext_vector_type(16))) __bf16 bf16x16;
typedef __attribute__((ext_vector_type(8)))  float  f32x8;
typedef __attribute__((ext_vector_type(4)))  unsigned int u32x4;
typedef __attribute__((ext_vector_type(8)))  int    i32x8;
typedef __attribute__((ext_vector_type(4)))  int    i32x4;

union FragA { uint32_t u[8]; bf16x16 v; };

__device__ __forceinline__ uint16_t f2bf(float f) {
  uint32_t u = __float_as_uint(f);
  u += 0x7FFFu + ((u >> 16) & 1u);       // round-to-nearest-even
  return (uint16_t)(u >> 16);
}
__device__ __forceinline__ uint32_t pack2bf(float a, float b) {
  return (uint32_t)f2bf(a) | ((uint32_t)f2bf(b) << 16);
}
// 16-bit WMMA A/B fragment K-pair offset for element jj (0..7), lane K-half kh.
__device__ __forceinline__ int koff(int jj, int kh) {
  return ((jj >> 2) << 4) + (kh << 3) + ((jj & 3) << 1);
}

static constexpr int Bsz  = 2;
static constexpr int Tlen = 2048;
static constexpr int Cdim = 4096;
static constexpr int G    = 8;
static constexpr int QPK  = 4;        // q heads per kv group
static constexpr int D    = 128;
static constexpr int SLOT = 48;       // (4 q + 1 k + 1 v) * 8 groups, 128 each
static constexpr int QKVN = SLOT * D; // 6144

#if defined(__has_builtin)
#  if __has_builtin(__builtin_amdgcn_tensor_load_to_lds)
#    define USE_TDM 1
#  endif
#endif

// ---------------------------------------------------------------------------
// TDM: load a [128 x 32] bf16 tile of a rows-row tensor (row stride = rowStride elements) from
// global into LDS with a 16-byte pad after each 64-byte row (stride 40 elems).
// D# per CDNA5 ISA ch.8: group0 = {flags, lds_addr, gaddr}, group1 = dims.
// ---------------------------------------------------------------------------
#ifdef USE_TDM
__device__ __forceinline__ void tdm_load_panel(uint32_t lds_addr, uint64_t gaddr,
                                               int rows, int rowStride) {
  u32x4 g0;
  g0[0] = 1u;                                          // count=1, user mode
  g0[1] = lds_addr;                                    // LDS byte address
  g0[2] = (uint32_t)gaddr;                             // global addr [31:0]
  g0[3] = ((uint32_t)(gaddr >> 32) & 0x01FFFFFFu) | (2u << 30); // [56:32]|type=2
  i32x8 g1;
  // data_size=1 (2B) | pad_enable | pad_interval=3 (16 DW = 64B) | pad_amount=3 (4 DW = 16B)
  g1[0] = (1 << 16) | (1 << 20) | (3 << 22) | (3 << 25);
  g1[1] = (rowStride & 0xFFFF) << 16;                  // tensor_dim0[15:0]
  g1[2] = ((rowStride >> 16) & 0xFFFF) | ((rows & 0xFFFF) << 16); // dim0 hi | dim1 lo
  g1[3] = ((rows >> 16) & 0xFFFF) | (32 << 16);        // dim1 hi | tile_dim0=32
  g1[4] = 128;                                         // tile_dim1=128, tile_dim2=0
  g1[5] = rowStride;                                   // tensor_dim0_stride[31:0]
  g1[6] = 0;                                           // stride0 hi | stride1 lo
  g1[7] = 0;
  i32x4 z4; z4[0] = 0; z4[1] = 0; z4[2] = 0; z4[3] = 0;
  i32x8 z8;
  #pragma unroll
  for (int i = 0; i < 8; ++i) z8[i] = 0;
  __builtin_amdgcn_tensor_load_to_lds(g0, g1, z4, z4, z8, 0);
}
#endif

// ---------------------------------------------------------------------------
// fp32 -> bf16 bulk conversion (4 elems / thread)
// ---------------------------------------------------------------------------
__global__ void cvt_bf16_kernel(const float* __restrict__ src,
                                uint16_t* __restrict__ dst, size_t n) {
  const size_t i = ((size_t)blockIdx.x * blockDim.x + threadIdx.x) * 4;
  if (i >= n) return;
  float4 v = *(const float4*)(src + i);
  uint2 o;
  o.x = pack2bf(v.x, v.y);
  o.y = pack2bf(v.z, v.w);
  *(uint2*)(dst + i) = o;
}

// ---------------------------------------------------------------------------
// GEMM: out[m,n] = sum_k A[m,k] * W[n,k]   (bf16 operands, fp32 out)
// 128x128 tile / 256-thread block; panels staged by double-buffered TDM so
// the next panel's DMA overlaps this panel's 64 WMMAs; 8 waves x 8 WMMA/step.
// ---------------------------------------------------------------------------
static constexpr int LDT   = 40;           // padded LDS row stride (elements)
static constexpr int PANEL = 128 * LDT;    // one panel, elements

__global__ __launch_bounds__(256) void gemm_tn_bf16(
    const uint16_t* __restrict__ A, const uint16_t* __restrict__ W,
    float* __restrict__ out, int M, int N, int K)
{
  __shared__ uint16_t As[2 * PANEL];
  __shared__ uint16_t Ws[2 * PANEL];

  const int tid  = threadIdx.x;
  const int wave = tid >> 5, lane = tid & 31;
  const int lm = lane & 15, kh = lane >> 4;
  const int wm = wave & 3;            // wave grid: 4 along M
  const int wn = wave >> 2;           //            2 along N
  const int rowBase = blockIdx.y * 128;
  const int colBase = blockIdx.x * 128;

#ifdef USE_TDM
  const uint64_t aBase = (uint64_t)(uintptr_t)A + (uint64_t)rowBase * K * 2;
  const uint64_t wBase = (uint64_t)(uintptr_t)W + (uint64_t)colBase * K * 2;
  const uint32_t ldsA0 = (uint32_t)(uintptr_t)&As[0];
  const uint32_t ldsW0 = (uint32_t)(uintptr_t)&Ws[0];
  // prologue: kick off panel 0 (TENSORcnt = 2 on wave 0)
  if (wave == 0) {
    tdm_load_panel(ldsA0, aBase, M, K);
    tdm_load_panel(ldsW0, wBase, N, K);
  }
#else
  const int ldr = tid >> 1;           // staging fallback: 2 threads per row
  const int ldc = (tid & 1) * 16;
  const uint16_t* Arow = A + (size_t)(rowBase + ldr) * K + ldc;
  const uint16_t* Wrow = W + (size_t)(colBase + ldr) * K + ldc;
#endif

  f32x8 acc[2][4];
  #pragma unroll
  for (int mt = 0; mt < 2; ++mt)
    #pragma unroll
    for (int nt = 0; nt < 4; ++nt)
      #pragma unroll
      for (int e = 0; e < 8; ++e) acc[mt][nt][e] = 0.0f;

  for (int k0 = 0; k0 < K; k0 += 32) {
    const int cur = (k0 >> 5) & 1;
#ifdef USE_TDM
    if (wave == 0) {
      if (k0 + 32 < K) {
        // issue next panel into the other buffer, then wait for the current
        // panel (<=2 outstanding leaves only the just-issued pair in flight;
        // TDM ops of one wave complete in order)
        const int nxt = cur ^ 1;
        tdm_load_panel(ldsA0 + (uint32_t)nxt * PANEL * 2,
                       aBase + (uint64_t)(k0 + 32) * 2, M, K);
        tdm_load_panel(ldsW0 + (uint32_t)nxt * PANEL * 2,
                       wBase + (uint64_t)(k0 + 32) * 2, N, K);
        __builtin_amdgcn_s_wait_tensorcnt(2);
      } else {
        __builtin_amdgcn_s_wait_tensorcnt(0);
      }
    }
#else
    *(uint4*)&As[cur * PANEL + ldr * LDT + ldc] = *(const uint4*)(Arow + k0);
    *(uint4*)&Ws[cur * PANEL + ldr * LDT + ldc] = *(const uint4*)(Wrow + k0);
#endif
    __syncthreads();

    const uint16_t* Ac = &As[cur * PANEL];
    const uint16_t* Wc = &Ws[cur * PANEL];
    FragA af[2], bfg[4];
    #pragma unroll
    for (int mt = 0; mt < 2; ++mt) {
      const int r = wm * 32 + mt * 16 + lm;
      #pragma unroll
      for (int jj = 0; jj < 8; ++jj)
        af[mt].u[jj] = *(const uint32_t*)&Ac[r * LDT + koff(jj, kh)];
    }
    #pragma unroll
    for (int nt = 0; nt < 4; ++nt) {
      const int r = wn * 64 + nt * 16 + lm;
      #pragma unroll
      for (int jj = 0; jj < 8; ++jj)
        bfg[nt].u[jj] = *(const uint32_t*)&Wc[r * LDT + koff(jj, kh)];
    }
    #pragma unroll
    for (int mt = 0; mt < 2; ++mt)
      #pragma unroll
      for (int nt = 0; nt < 4; ++nt)
        acc[mt][nt] = __builtin_amdgcn_wmma_f32_16x16x32_bf16(
            false, af[mt].v, false, bfg[nt].v, (short)0, acc[mt][nt], false, false);
    __syncthreads();   // all reads of buf[cur] done before TDM reuses it
  }

  #pragma unroll
  for (int mt = 0; mt < 2; ++mt)
    #pragma unroll
    for (int nt = 0; nt < 4; ++nt)
      #pragma unroll
      for (int r = 0; r < 8; ++r) {
        const int row = rowBase + wm * 32 + mt * 16 + r + 8 * kh;
        const int col = colBase + wn * 64 + nt * 16 + lm;
        out[(size_t)row * N + col] = acc[mt][nt][r];
      }
}

// ---------------------------------------------------------------------------
// RoPE + bf16 pack: reads fp32 qkv, writes bf16 qkvh.
// q slots (g*6+0..3) and k slot (g*6+4) rotated; v slot (g*6+5) converted.
// One thread per (b,t,slot,d-pair).
// ---------------------------------------------------------------------------
__global__ void rope_kernel(const float* __restrict__ qkv,
                            uint16_t* __restrict__ qkvh,
                            const float* __restrict__ cosT,
                            const float* __restrict__ sinT)
{
  const size_t idx = (size_t)blockIdx.x * blockDim.x + threadIdx.x;
  const int d    = (int)(idx & 63);
  size_t rest    = idx >> 6;
  const int slot = (int)(rest % SLOT); rest /= SLOT;
  const int t    = (int)(rest % Tlen);
  const int b    = (int)(rest / Tlen);
  if (b >= Bsz) return;

  const size_t base = ((size_t)(b * Tlen + t) * SLOT + slot) * D;
  const float t1 = qkv[base + d];
  const float t2 = qkv[base + 64 + d];
  float o1 = t1, o2 = t2;
  if ((slot % 6) < 5) {              // q or k: rotate
    const float c = cosT[t * 64 + d];
    const float s = sinT[t * 64 + d];
    o1 = t1 * c - t2 * s;
    o2 = t1 * s + t2 * c;
  }
  qkvh[base + d]      = f2bf(o1);
  qkvh[base + 64 + d] = f2bf(o2);
}

// ---------------------------------------------------------------------------
// Flash-attention: block = (qtile16, group, batch); 4 waves = 4 q-heads
// sharing the KV group.  bf16 K/V staged 32 keys at a time in LDS.
// ---------------------------------------------------------------------------
static constexpr int KLD = 136;   // K_lds row stride (keys x d)
static constexpr int VLD = 40;    // Vt_lds row stride (d x keys)
static constexpr int PLD = 40;    // P_lds row stride

__global__ __launch_bounds__(128) void attn_kernel(
    const uint16_t* __restrict__ qkvh, uint16_t* __restrict__ yh)
{
  __shared__ uint16_t Ks[32 * KLD];
  __shared__ uint16_t Vs[128 * VLD];
  __shared__ uint16_t Ps[4 * 16 * PLD];

  const int qtile = blockIdx.x;
  const int g     = blockIdx.y;
  const int b     = blockIdx.z;
  const int tid   = threadIdx.x;
  const int wv    = tid >> 5;          // q-head within group
  const int lane  = tid & 31;
  const int lm    = lane & 15, kh = lane >> 4;

  // --- Q fragments (16 rows x 128 d as 4 K-steps of 32), straight bf16 loads
  FragA qf[4];
  {
    const int qrow = qtile * 16 + lm;
    const uint16_t* qp = qkvh + ((size_t)(b * Tlen + qrow) * SLOT + g * 6 + wv) * D;
    #pragma unroll
    for (int s = 0; s < 4; ++s)
      #pragma unroll
      for (int jj = 0; jj < 8; ++jj)
        qf[s].u[jj] = *(const uint32_t*)&qp[s * 32 + koff(jj, kh)];
  }

  f32x8 acc[8];
  float mprev[8], lsum[8];
  #pragma unroll
  for (int nt = 0; nt < 8; ++nt)
    #pragma unroll
    for (int e = 0; e < 8; ++e) acc[nt][e] = 0.0f;
  #pragma unroll
  for (int r = 0; r < 8; ++r) { mprev[r] = -1e30f; lsum[r] = 0.0f; }

  const int qlast = qtile * 16 + 15;
  const int nch   = (qlast + 32) / 32;    // ceil((qlast+1)/32)
  const float scl = 0.08838834764831845f; // 1/sqrt(128)

  for (int ch = 0; ch < nch; ++ch) {
    const int j0 = ch * 32;
    // --- cooperative K/V staging (bf16 copies; V transposed)
    for (int i = tid; i < 32 * 128; i += 128) {
      const int key = i >> 7, d = i & 127;
      const size_t kb = ((size_t)(b * Tlen + j0 + key) * SLOT + g * 6 + 4) * D + d;
      Ks[key * KLD + d] = qkvh[kb];        // K: [key][d]
      Vs[d * VLD + key] = qkvh[kb + D];    // V transposed: [d][key]
    }
    __syncthreads();

    // --- S = Q K^T  (16 x 32 scores, two 16-col WMMA tiles, 4 K-steps)
    f32x8 s0, s1;
    #pragma unroll
    for (int e = 0; e < 8; ++e) { s0[e] = 0.0f; s1[e] = 0.0f; }
    #pragma unroll
    for (int s = 0; s < 4; ++s) {
      FragA bk0, bk1;
      #pragma unroll
      for (int jj = 0; jj < 8; ++jj) {
        const int k = s * 32 + koff(jj, kh);
        bk0.u[jj] = *(const uint32_t*)&Ks[lm * KLD + k];
        bk1.u[jj] = *(const uint32_t*)&Ks[(16 + lm) * KLD + k];
      }
      s0 = __builtin_amdgcn_wmma_f32_16x16x32_bf16(false, qf[s].v, false, bk0.v,
                                                   (short)0, s0, false, false);
      s1 = __builtin_amdgcn_wmma_f32_16x16x32_bf16(false, qf[s].v, false, bk1.v,
                                                   (short)0, s1, false, false);
    }

    // --- scale + causal mask + online softmax update
    float alpha[8];
    #pragma unroll
    for (int r = 0; r < 8; ++r) {
      const int m    = r + 8 * kh;
      const int qpos = qtile * 16 + m;
      const int c0 = j0 + lm, c1 = j0 + 16 + lm;
      const float v0 = (c0 <= qpos) ? s0[r] * scl : -1e30f;
      const float v1 = (c1 <= qpos) ? s1[r] * scl : -1e30f;
      float rm = fmaxf(v0, v1);
      rm = fmaxf(rm, __shfl_xor(rm, 8, 16));
      rm = fmaxf(rm, __shfl_xor(rm, 4, 16));
      rm = fmaxf(rm, __shfl_xor(rm, 2, 16));
      rm = fmaxf(rm, __shfl_xor(rm, 1, 16));
      const float mnew = fmaxf(mprev[r], rm);
      alpha[r] = __expf(mprev[r] - mnew);
      const float p0 = __expf(v0 - mnew);
      const float p1 = __expf(v1 - mnew);
      float ps = p0 + p1;
      ps += __shfl_xor(ps, 8, 16);
      ps += __shfl_xor(ps, 4, 16);
      ps += __shfl_xor(ps, 2, 16);
      ps += __shfl_xor(ps, 1, 16);
      lsum[r] = lsum[r] * alpha[r] + ps;
      mprev[r] = mnew;
      // P tile (C-layout) -> LDS row-major bf16, for A-layout re-read
      Ps[wv * 16 * PLD + m * PLD + lm]      = f2bf(p0);
      Ps[wv * 16 * PLD + m * PLD + 16 + lm] = f2bf(p1);
    }
    #pragma unroll
    for (int nt = 0; nt < 8; ++nt)
      #pragma unroll
      for (int r = 0; r < 8; ++r) acc[nt][r] *= alpha[r];

    // --- O += P (16x32) x V (32x128): A-frag from Ps, B-frags from Vt
    FragA pf;
    #pragma unroll
    for (int jj = 0; jj < 8; ++jj)
      pf.u[jj] = *(const uint32_t*)&Ps[wv * 16 * PLD + lm * PLD + koff(jj, kh)];
    #pragma unroll
    for (int nt = 0; nt < 8; ++nt) {
      FragA bv;
      #pragma unroll
      for (int jj = 0; jj < 8; ++jj)
        bv.u[jj] = *(const uint32_t*)&Vs[(nt * 16 + lm) * VLD + koff(jj, kh)];
      acc[nt] = __builtin_amdgcn_wmma_f32_16x16x32_bf16(false, pf.v, false, bv.v,
                                                        (short)0, acc[nt], false, false);
    }
    __syncthreads();   // before next chunk overwrites Ks/Vs
  }

  // --- epilogue: normalize, write bf16 y[b][t][g*4+wv][d]
  #pragma unroll
  for (int r = 0; r < 8; ++r) lsum[r] = 1.0f / lsum[r];
  #pragma unroll
  for (int nt = 0; nt < 8; ++nt)
    #pragma unroll
    for (int r = 0; r < 8; ++r) {
      const int m = r + 8 * kh;
      const int d = nt * 16 + lm;
      yh[((size_t)(b * Tlen + qtile * 16 + m) * (G * QPK) + g * QPK + wv) * D + d] =
          f2bf(acc[nt][r] * lsum[r]);
    }
}

// ---------------------------------------------------------------------------
// Launch: bf16 pre-convert -> qkv GEMM -> RoPE(+pack) -> attention -> proj
// Workspace (bytes):
//   qkv  fp32 : 100,663,296
//   qkvh bf16 :  50,331,648
//   xh   bf16 :  33,554,432
//   Wah  bf16 :  50,331,648
//   Wph  bf16 :  33,554,432
//   yh   bf16 :  33,554,432    total ~302 MB
// ---------------------------------------------------------------------------
extern "C" void kernel_launch(void* const* d_in, const int* in_sizes, int n_in,
                              void* d_out, int out_size, void* d_ws, size_t ws_size,
                              hipStream_t stream) {
  const float* x      = (const float*)d_in[0];
  const float* cosT   = (const float*)d_in[1];
  const float* sinT   = (const float*)d_in[2];
  const float* W_attn = (const float*)d_in[3];
  const float* W_proj = (const float*)d_in[4];

  const int M = Bsz * Tlen;                  // 4096
  const size_t nX  = (size_t)M * Cdim;       // 16.7M
  const size_t nWa = (size_t)QKVN * Cdim;    // 25.2M
  const size_t nWp = (size_t)Cdim * Cdim;    // 16.7M
  const size_t nQKV = (size_t)M * QKVN;      // 25.2M

  char* ws = (char*)d_ws;
  float*    qkv  = (float*)ws;                 ws += nQKV * 4;
  uint16_t* qkvh = (uint16_t*)ws;              ws += nQKV * 2;
  uint16_t* xh   = (uint16_t*)ws;              ws += nX * 2;
  uint16_t* Wah  = (uint16_t*)ws;              ws += nWa * 2;
  uint16_t* Wph  = (uint16_t*)ws;              ws += nWp * 2;
  uint16_t* yh   = (uint16_t*)ws;
  float* out = (float*)d_out;

  // 0) bf16 pre-conversion of all GEMM operands (once, not per tile)
  cvt_bf16_kernel<<<(unsigned)(nX  / 4 / 256), 256, 0, stream>>>(x,      xh,  nX);
  cvt_bf16_kernel<<<(unsigned)(nWa / 4 / 256), 256, 0, stream>>>(W_attn, Wah, nWa);
  cvt_bf16_kernel<<<(unsigned)(nWp / 4 / 256), 256, 0, stream>>>(W_proj, Wph, nWp);

  // 1) QKV projection: [4096 x 4096] x [6144 x 4096]^T -> fp32 qkv
  gemm_tn_bf16<<<dim3(QKVN / 128, M / 128), 256, 0, stream>>>(
      xh, Wah, qkv, M, QKVN, Cdim);

  // 2) RoPE + pack to bf16 qkvh
  {
    const size_t total = (size_t)Bsz * Tlen * SLOT * 64;
    rope_kernel<<<(unsigned)(total / 256), 256, 0, stream>>>(qkv, qkvh, cosT, sinT);
  }

  // 3) Flash attention -> bf16 yh
  attn_kernel<<<dim3(Tlen / 16, G, Bsz), 128, 0, stream>>>(qkvh, yh);

  // 4) Output projection: [4096 x 4096] x [4096 x 4096]^T -> fp32 out
  gemm_tn_bf16<<<dim3(Cdim / 128, M / 128), 256, 0, stream>>>(
      yh, Wph, out, M, Cdim, Cdim);
}
